// GAT_my_attention_64372969832704
// MI455X (gfx1250) — compile-verified, matrix operand order checked by reading, not run on hardware
//
#include <hip/hip_runtime.h>
#include <hip/hip_bf16.h>

// GAT (2 layers, sign-based attention) for MI455X / gfx1250.
// GEMMs use v_wmma_f32_16x16x32_bf16 (f32 accumulate). Segment softmax is
// computed EXACTLY via integer counters (alpha is +-T only), avoiding float
// segment reductions. Scatter-add uses global f32 atomics (L2-resident).
//
// Workspace layout (~67 MB): xb(bf16 N*64) | h1(f32 N*64) | out1(f32 N*64)
//                            signs(E') | nPos(N) | deg(N) | w1b | w2b

typedef __attribute__((ext_vector_type(16))) __bf16 v16bf;
typedef __attribute__((ext_vector_type(8)))  float  v8f;

#define ENEG2 0.13533528323661270f /* exp(-2*T), T = 1.0 */

__device__ __forceinline__ unsigned short f32_to_bf16_rne(float f) {
  unsigned int u = __float_as_uint(f);
  u += 0x7FFFu + ((u >> 16) & 1u);   // round-to-nearest-even
  return (unsigned short)(u >> 16);
}

__global__ void cvt_f32_bf16_kernel(const float* __restrict__ in,
                                    unsigned short* __restrict__ out, int n) {
  int i = blockIdx.x * blockDim.x + threadIdx.x;
  if (i < n) out[i] = f32_to_bf16_rne(in[i]);
}

// K-pair start for VGPR slot v in the 16-bit A/B 16x32 layout (ISA 7.12.2):
// half 0: v0..3 -> K=0..7,  v4..7 -> K=16..23
// half 1: v0..3 -> K=8..15, v4..7 -> K=24..31
__device__ __forceinline__ int wmma_kpair(int v, int half) {
  return 2 * v + (half ? ((v < 4) ? 8 : 16) : ((v < 4) ? 0 : 8));
}

// C[M x (NT*16)] = A[M x K](bf16) * Wt[(NT*16) x K](bf16, torch [n][k] layout)^T
// One wave per 16-row tile; NT n-tiles; K multiple of 32.
template <int NT>
__global__ void wmma_gemm_kernel(const unsigned short* __restrict__ A,
                                 const unsigned short* __restrict__ Wt,
                                 float* __restrict__ C, int Mtiles, int K) {
  const int wave = blockIdx.x * (blockDim.x >> 5) + (threadIdx.x >> 5);
  if (wave >= Mtiles) return;
  const int lane = threadIdx.x & 31;
  const int half = lane >> 4;
  const int lr   = lane & 15;
  const int row0 = wave * 16;
  const int Ntot = NT * 16;

  v8f zero = {};
  v8f acc[NT];
#pragma unroll
  for (int t = 0; t < NT; ++t) acc[t] = zero;

  for (int k0 = 0; k0 < K; k0 += 32) {
    union { v16bf v; unsigned int u[8]; } a;
    const unsigned short* arow = A + (size_t)(row0 + lr) * K + k0;
#pragma unroll
    for (int v = 0; v < 8; ++v)
      a.u[v] = *(const unsigned int*)(arow + wmma_kpair(v, half));
#pragma unroll
    for (int t = 0; t < NT; ++t) {
      union { v16bf v; unsigned int u[8]; } b;
      const unsigned short* brow = Wt + (size_t)(t * 16 + lr) * K + k0;
#pragma unroll
      for (int v = 0; v < 8; ++v)
        b.u[v] = *(const unsigned int*)(brow + wmma_kpair(v, half));
      acc[t] = __builtin_amdgcn_wmma_f32_16x16x32_bf16(
          false, a.v, false, b.v, (short)0, acc[t], false, false);
    }
  }
  // 32-bit C/D 16x16 layout: VGPR v -> M = v + 8*half, lane%16 -> N
#pragma unroll
  for (int t = 0; t < NT; ++t)
#pragma unroll
    for (int v = 0; v < 8; ++v)
      C[(size_t)(row0 + v + 8 * half) * Ntot + t * 16 + lr] = acc[t][v];
}

// Scalar fallback for a row remainder (unused when N % 16 == 0).
__global__ void gemm_tail_kernel(const unsigned short* __restrict__ A,
                                 const unsigned short* __restrict__ Wt,
                                 float* __restrict__ C, int rowStart, int rows,
                                 int Ntot, int K) {
  int i = blockIdx.x * blockDim.x + threadIdx.x;
  if (i >= rows * Ntot) return;
  int r = rowStart + i / Ntot, n = i % Ntot;
  float s = 0.f;
  for (int k = 0; k < K; ++k) {
    float a = __uint_as_float((unsigned int)A[(size_t)r * K + k] << 16);
    float b = __uint_as_float((unsigned int)Wt[(size_t)n * K + k] << 16);
    s += a * b;
  }
  C[(size_t)r * Ntot + n] = s;
}

// Per-edge: sign(<h[dst], h[src]>), count positives (and degree) per src.
// Edges e<E come from edge_index; e>=E are self loops (src=dst=e-E).
template <int C, bool COUNT_DEG>
__global__ void edge_alpha_kernel(const int* __restrict__ ei, int E, int Np,
                                  const float* __restrict__ h,
                                  unsigned char* __restrict__ sgn,
                                  int* __restrict__ nPos, int* __restrict__ deg) {
  int e = blockIdx.x * blockDim.x + threadIdx.x;
  if (e >= E + Np) return;
  int s, d;
  if (e < E) { s = ei[e]; d = ei[E + e]; } else { s = d = e - E; }
  const float4* hs = (const float4*)(h + (size_t)s * C);
  const float4* hd = (const float4*)(h + (size_t)d * C);
  float dot = 0.f;
#pragma unroll
  for (int i = 0; i < C / 4; ++i) {
    float4 a = hs[i], b = hd[i];
    dot += a.x * b.x + a.y * b.y + a.z * b.z + a.w * b.w;
  }
  bool pos = dot > 0.0f;
  sgn[e] = pos ? 1 : 0;
  if (pos) atomicAdd(&nPos[s], 1);
  if (COUNT_DEG) atomicAdd(&deg[s], 1);
}

// out[dst] += h[src] * w, with exact segment-softmax weight from counters.
template <int C>
__global__ void edge_msg_kernel(const int* __restrict__ ei, int E, int Np,
                                const float* __restrict__ h,
                                const unsigned char* __restrict__ sgn,
                                const int* __restrict__ nPos,
                                const int* __restrict__ deg,
                                float* __restrict__ out) {
  long long tid = (long long)blockIdx.x * blockDim.x + threadIdx.x;
  if (tid >= (long long)(E + Np) * C) return;
  int e = (int)(tid / C);
  int c = (int)(tid % C);
  int s, d;
  if (e < E) { s = ei[e]; d = ei[E + e]; } else { s = d = e - E; }
  int np = nPos[s], dg = deg[s];
  float w;
  if (np > 0) {
    float denom = (float)np + (float)(dg - np) * ENEG2;
    w = (sgn[e] ? 1.0f : ENEG2) / denom;
  } else {
    w = 1.0f / (float)dg;  // all-negative segment: e=1 for every edge
  }
  atomicAdd(&out[(size_t)d * C + c], h[(size_t)s * C + c] * w);
}

__global__ void bias_relu_bf16_kernel(const float* __restrict__ in,
                                      const float* __restrict__ b,
                                      unsigned short* __restrict__ outbf,
                                      int n, int Cch) {
  int i = blockIdx.x * blockDim.x + threadIdx.x;
  if (i >= n) return;
  float v = in[i] + b[i % Cch];
  outbf[i] = f32_to_bf16_rne(v > 0.f ? v : 0.f);
}

// In-place: io[r,:] = log_softmax(io[r,:] + b2)
__global__ void bias_logsoftmax16_kernel(float* __restrict__ io,
                                         const float* __restrict__ b, int Np) {
  int r = blockIdx.x * blockDim.x + threadIdx.x;
  if (r >= Np) return;
  float v[16];
  float m = -1e30f;
#pragma unroll
  for (int c = 0; c < 16; ++c) {
    v[c] = io[(size_t)r * 16 + c] + b[c];
    m = fmaxf(m, v[c]);
  }
  float s = 0.f;
#pragma unroll
  for (int c = 0; c < 16; ++c) s += expf(v[c] - m);
  float ls = logf(s);
#pragma unroll
  for (int c = 0; c < 16; ++c) io[(size_t)r * 16 + c] = v[c] - m - ls;
}

static inline int divUp(long long a, long long b) { return (int)((a + b - 1) / b); }

extern "C" void kernel_launch(void* const* d_in, const int* in_sizes, int n_in,
                              void* d_out, int out_size, void* d_ws, size_t ws_size,
                              hipStream_t stream) {
  (void)n_in; (void)out_size; (void)ws_size;
  const float* x  = (const float*)d_in[0];
  const int*   ei = (const int*)d_in[1];   // [2, E] int32
  const float* W1 = (const float*)d_in[2]; // [64, 64] torch layout [out][in]
  const float* b1 = (const float*)d_in[3];
  const float* W2 = (const float*)d_in[4]; // [16, 64]
  const float* b2 = (const float*)d_in[5];

  const int Np = in_sizes[0] / 64;
  const int E  = in_sizes[1] / 2;
  const int Ep = E + Np;

  // Carve workspace (256B-aligned slices).
  char* ws = (char*)d_ws;
  size_t off = 0;
  auto carve = [&](size_t bytes) -> char* {
    char* p = ws + off;
    off += (bytes + 255) & ~(size_t)255;
    return p;
  };
  unsigned short* xb   = (unsigned short*)carve((size_t)Np * 64 * 2); // x bf16; later relu(h1) bf16
  float*          h1   = (float*)carve((size_t)Np * 64 * 4);          // pre-attn feats; later h2
  float*          out1 = (float*)carve((size_t)Np * 64 * 4);
  unsigned char*  sgn  = (unsigned char*)carve((size_t)Ep);
  int*            nPos = (int*)carve((size_t)Np * 4);
  int*            deg  = (int*)carve((size_t)Np * 4);
  unsigned short* w1b  = (unsigned short*)carve(64 * 64 * 2);
  unsigned short* w2b  = (unsigned short*)carve(16 * 64 * 2);

  // --- inputs -> bf16 for WMMA ---
  cvt_f32_bf16_kernel<<<divUp((long long)Np * 64, 256), 256, 0, stream>>>(x, xb, Np * 64);
  cvt_f32_bf16_kernel<<<divUp(64 * 64, 256), 256, 0, stream>>>(W1, w1b, 64 * 64);
  cvt_f32_bf16_kernel<<<divUp(16 * 64, 256), 256, 0, stream>>>(W2, w2b, 16 * 64);

  const int Mtiles = Np / 16;
  const int tail = Np - Mtiles * 16;

  // --- layer 1: h1 = x @ W1^T (WMMA bf16 -> f32) ---
  wmma_gemm_kernel<4><<<divUp(Mtiles, 8), 256, 0, stream>>>(xb, w1b, h1, Mtiles, 64);
  if (tail)
    gemm_tail_kernel<<<divUp((long long)tail * 64, 256), 256, 0, stream>>>(
        xb, w1b, h1, Mtiles * 16, tail, 64, 64);

  hipMemsetAsync(nPos, 0, (size_t)Np * 4, stream);
  hipMemsetAsync(deg, 0, (size_t)Np * 4, stream);
  edge_alpha_kernel<64, true><<<divUp(Ep, 256), 256, 0, stream>>>(ei, E, Np, h1, sgn, nPos, deg);
  hipMemsetAsync(out1, 0, (size_t)Np * 64 * 4, stream);
  edge_msg_kernel<64><<<divUp((long long)Ep * 64, 256), 256, 0, stream>>>(
      ei, E, Np, h1, sgn, nPos, deg, out1);
  bias_relu_bf16_kernel<<<divUp((long long)Np * 64, 256), 256, 0, stream>>>(
      out1, b1, xb, Np * 64, 64);  // xb now holds relu(h1+b1) in bf16

  // --- layer 2: h2 = relu(h1) @ W2^T ---
  float* h2 = h1;  // reuse
  wmma_gemm_kernel<1><<<divUp(Mtiles, 8), 256, 0, stream>>>(xb, w2b, h2, Mtiles, 64);
  if (tail)
    gemm_tail_kernel<<<divUp((long long)tail * 16, 256), 256, 0, stream>>>(
        xb, w2b, h2, Mtiles * 16, tail, 16, 64);

  hipMemsetAsync(nPos, 0, (size_t)Np * 4, stream);
  edge_alpha_kernel<16, false><<<divUp(Ep, 256), 256, 0, stream>>>(ei, E, Np, h2, sgn, nPos, deg);

  float* out2 = (float*)d_out;
  hipMemsetAsync(out2, 0, (size_t)Np * 16 * 4, stream);
  edge_msg_kernel<16><<<divUp((long long)Ep * 16, 256), 256, 0, stream>>>(
      ei, E, Np, h2, sgn, nPos, deg, out2);
  bias_logsoftmax16_kernel<<<divUp(Np, 256), 256, 0, stream>>>(out2, b2, Np);
}
